// VectorQuantizer_kmeans_9981503995961
// MI455X (gfx1250) — matrix operand order repaired and checked by reading, not compiled, not run on hardware
//
#include <hip/hip_runtime.h>

// ---------------------------------------------------------------------------
// VQ-VAE vector quantizer for MI455X (gfx1250, wave32, WMMA + async LDS DMA).
// d[b,n] = ||z_b||^2 + ||e_n||^2 - 2 z_b . e_n ; argmin_n ; z_q = emb[idx]
// ||z||^2 is row-constant -> argmin over (e_norm[n] - 2 * dot(z,e_n)).
// Distance GEMM in bf16 WMMA (f32 accum); outputs computed exactly in f32.
// Codebook panels double-buffered in LDS via global_load_async_to_lds_b128.
// ---------------------------------------------------------------------------

#define VQ_BATCH 131072
#define VQ_NE    1024
#define VQ_DIM   256
#define VQ_BETA  0.25f

// LDS panel geometry: 64 codes/panel, rows padded 256->264 bf16 (528 B) so
// consecutive codes land 132 LDS words apart -> lane l16 hits bank 4*l16
// (conflict-free ds_load_b128 across the 16-lane half).
#define VQ_PANEL_CODES 64
#define VQ_N_PANELS    (VQ_NE / VQ_PANEL_CODES)   // 16
#define VQ_LROW        264                         // padded row, bf16 elements
#define VQ_PANEL_ELTS  (VQ_PANEL_CODES * VQ_LROW)  // 16896
#define VQ_LDS_BYTES   (2 * VQ_PANEL_ELTS * 2)     // 67584

#if defined(__has_builtin)
# if __has_builtin(__builtin_amdgcn_global_load_async_to_lds_b128)
#  define VQ_ASYNC 1
# endif
# if __has_builtin(__builtin_amdgcn_s_wait_asynccnt)
#  define VQ_WAIT_BUILTIN 1
# endif
#endif
#ifndef VQ_ASYNC
# define VQ_ASYNC 0
#endif
#ifndef VQ_WAIT_BUILTIN
# define VQ_WAIT_BUILTIN 0
#endif

typedef __attribute__((ext_vector_type(16))) __bf16 v16bf;
typedef __attribute__((ext_vector_type(8)))  float  v8f;
typedef __attribute__((ext_vector_type(4)))  int    v4i;

// Typed address-space pointers for the async DMA builtin:
// proto (from diagnostic): (v4i AS1* src, v4i AS3* dst, imm offset, imm cpol)
typedef __attribute__((address_space(1))) v4i vq_glob_v4i;
typedef __attribute__((address_space(3))) v4i vq_lds_v4i;

__device__ __forceinline__ void vq_wait_async() {
#if VQ_ASYNC
# if VQ_WAIT_BUILTIN
  __builtin_amdgcn_s_wait_asynccnt(0);
# else
  asm volatile("s_wait_asynccnt 0" ::: "memory");
# endif
#endif
}

// Stage one 64-code panel (64 x 512 B, row-major bf16) into padded LDS rows.
// Each wave copies exactly one 512 B row per j-step -> fully coalesced.
__device__ __forceinline__ void vq_stage_panel(const __bf16* __restrict__ embb,
                                               __bf16* __restrict__ lbuf,
                                               int panel, int tid) {
  const char* gbase = (const char*)embb + (size_t)panel * (VQ_PANEL_CODES * VQ_DIM * 2);
  char* lbase = (char*)lbuf;
#pragma unroll
  for (int j = 0; j < 8; ++j) {
    const int i = tid + j * 256;        // chunk id: 2048 x 16 B per panel
    const int r = i >> 5;               // code row in panel
    const int o = (i & 31) << 4;        // 16 B chunk within row
    const char* src = gbase + r * (VQ_DIM * 2) + o;
    char*       dst = lbase + r * (VQ_LROW * 2) + o;
#if VQ_ASYNC
    __builtin_amdgcn_global_load_async_to_lds_b128(
        (vq_glob_v4i*)(src), (vq_lds_v4i*)(dst), 0, 0);
#else
    *(uint4*)dst = *(const uint4*)src;
#endif
  }
}

// ---------------- init: zero histogram + loss accumulator ----------------
__global__ __launch_bounds__(256) void vq_init_kernel(int* __restrict__ counts,
                                                      float* __restrict__ loss_acc) {
  const int i = blockIdx.x * blockDim.x + threadIdx.x;
  if (i < VQ_NE) counts[i] = 0;
  if (i == 0) loss_acc[0] = 0.0f;
}

// ------------- prep: emb f32 -> bf16 (row-major) + e_norm[n] -------------
__global__ __launch_bounds__(256) void vq_prep_kernel(const float* __restrict__ emb,
                                                      __bf16* __restrict__ embb,
                                                      float* __restrict__ enorm) {
  __shared__ float red[8];
  const int n = blockIdx.x;     // code index
  const int t = threadIdx.x;    // dim index
  const float v = emb[(size_t)n * VQ_DIM + t];
  embb[(size_t)n * VQ_DIM + t] = (__bf16)v;
  float s = v * v;
#pragma unroll
  for (int off = 16; off >= 1; off >>= 1) s += __shfl_xor(s, off, 32);
  if ((t & 31) == 0) red[t >> 5] = s;
  __syncthreads();
  if (t < 8) {
    float s2 = red[t];
#pragma unroll
    for (int off = 4; off >= 1; off >>= 1) s2 += __shfl_xor(s2, off, 32);
    if (t == 0) enorm[n] = s2;
  }
}

// ---------------- distance GEMM + argmin via v_wmma bf16 -----------------
// One wave owns a 16-row M-tile; 8 waves/block. The block double-buffers
// 64-code panels of the bf16 codebook in LDS (async DMA overlapped with
// WMMA compute on the previous panel).
__global__ __launch_bounds__(256)
void vq_argmin_kernel(const float* __restrict__ z,
                      const __bf16* __restrict__ embb,
                      const float* __restrict__ enorm,
                      int* __restrict__ idx_ws,
                      int* __restrict__ counts) {
  extern __shared__ __bf16 vq_lds[];    // 2 * VQ_PANEL_ELTS
  const int tid  = threadIdx.x;
  const int lane = tid & 31;
  const int wave = tid >> 5;            // 0..7
  const int half = lane >> 4;           // 0: lanes 0-15, 1: lanes 16-31
  const int l16  = lane & 15;
  const int row0 = (blockIdx.x * 8 + wave) * 16;

  // Kick off panel-0 DMA before the (long) A-tile load/convert prologue.
  vq_stage_panel(embb, vq_lds, 0, tid);

  // ---- load + convert A tile (WMMA 16-bit A layout) ----
  // lane L<16 : row M=L,    elems 0-7 -> K=32c+0..7,  elems 8-15 -> K=32c+16..23
  // lane L>=16: row M=L-16, elems 0-7 -> K=32c+8..15, elems 8-15 -> K=32c+24..31
  const float* __restrict__ zrow = z + (size_t)(row0 + l16) * VQ_DIM;
  v16bf a[8];
#pragma unroll
  for (int c = 0; c < 8; ++c) {
    const int kb = 32 * c + half * 8;
    const float4 f0 = *(const float4*)(zrow + kb);
    const float4 f1 = *(const float4*)(zrow + kb + 4);
    const float4 f2 = *(const float4*)(zrow + kb + 16);
    const float4 f3 = *(const float4*)(zrow + kb + 20);
    const float ft[16] = {f0.x, f0.y, f0.z, f0.w, f1.x, f1.y, f1.z, f1.w,
                          f2.x, f2.y, f2.z, f2.w, f3.x, f3.y, f3.z, f3.w};
    v16bf av;
#pragma unroll
    for (int e = 0; e < 16; ++e) av[e] = (__bf16)ft[e];
    a[c] = av;
  }

  float best[8];
  int   bidx[8];
#pragma unroll
  for (int v = 0; v < 8; ++v) { best[v] = 3.0e38f; bidx[v] = 0; }

  vq_wait_async();
  __syncthreads();                      // panel 0 resident

  for (int p = 0; p < VQ_N_PANELS; ++p) {
    const int cur = p & 1;
    if (p + 1 < VQ_N_PANELS)            // uniform branch: EXEC stays full
      vq_stage_panel(embb, vq_lds + (cur ^ 1) * VQ_PANEL_ELTS, p + 1, tid);

    const __bf16* __restrict__ lbuf = vq_lds + cur * VQ_PANEL_ELTS;
#pragma unroll
    for (int nt = 0; nt < 4; ++nt) {
      // B layout (32x16 bf16), matching A: elems 0-7 -> K 32c+8*half+0..7,
      // elems 8-15 -> K 32c+16+8*half+0..7.
      const __bf16* brow = lbuf + (nt * 16 + l16) * VQ_LROW + half * 8;
      v8f acc = {};
#pragma unroll
      for (int c = 0; c < 8; ++c) {
        typedef __attribute__((ext_vector_type(8))) __bf16 v8bf_t;
        const v8bf_t b0 = *(const v8bf_t*)(brow + 32 * c);
        const v8bf_t b1 = *(const v8bf_t*)(brow + 32 * c + 16);
        const v16bf b = __builtin_shufflevector(b0, b1, 0, 1, 2, 3, 4, 5, 6, 7, 8, 9, 10, 11, 12, 13, 14, 15);
        acc = __builtin_amdgcn_wmma_f32_16x16x32_bf16(
            /*neg_a=*/false, a[c], /*neg_b=*/false, b,
            /*c_mod=*/(short)0, acc, /*reuse_a=*/false, /*reuse_b=*/false);
      }
      const int   nn = p * VQ_PANEL_CODES + nt * 16 + l16;
      const float en = enorm[nn];
#pragma unroll
      for (int v = 0; v < 8; ++v) {
        const float dv = en - 2.0f * acc[v];  // + ||z||^2 (row const, dropped)
        const bool lt = dv < best[v];         // strict: keeps earliest n per lane
        best[v] = lt ? dv : best[v];
        bidx[v] = lt ? nn : bidx[v];
      }
    }
    vq_wait_async();                    // next panel DMA done (overlapped)
    __syncthreads();                    // all waves done reading/writing LDS
  }

  // ---- cross-lane argmin within each 16-lane half (D row = v + 8*half) ----
#pragma unroll
  for (int v = 0; v < 8; ++v) {
    float bv = best[v];
    int   bi = bidx[v];
#pragma unroll
    for (int off = 1; off < 16; off <<= 1) {
      const float ov = __shfl_xor(bv, off, 32);
      const int   oi = __shfl_xor(bi, off, 32);
      const bool take = (ov < bv) || (ov == bv && oi < bi);  // argmin: lowest idx
      bv = take ? ov : bv;
      bi = take ? oi : bi;
    }
    if (l16 == 0) {
      const int row = row0 + v + 8 * half;
      idx_ws[row] = bi;
      atomicAdd(&counts[bi], 1);
    }
  }
}

// --------- gather z_q = emb[idx] (exact f32), loss partial, idx out ---------
__global__ __launch_bounds__(256)
void vq_gather_kernel(const float* __restrict__ z,
                      const float* __restrict__ emb,
                      const int* __restrict__ idx_ws,
                      float* __restrict__ zq_out,
                      float* __restrict__ idxf_out,
                      float* __restrict__ loss_acc) {
  __shared__ float red[8];
  const int row = blockIdx.x;
  const int t   = threadIdx.x;
  const int id  = idx_ws[row];
  const float e  = emb[(size_t)id * VQ_DIM + t];
  const float zv = z[(size_t)row * VQ_DIM + t];
  zq_out[(size_t)row * VQ_DIM + t] = e;  // z + sg(z_q - z) == z_q forward
  float d = e - zv;
  d = d * d;
#pragma unroll
  for (int off = 16; off >= 1; off >>= 1) d += __shfl_xor(d, off, 32);
  if ((t & 31) == 0) red[t >> 5] = d;
  __syncthreads();
  if (t < 8) {
    float s = red[t];
#pragma unroll
    for (int off = 4; off >= 1; off >>= 1) s += __shfl_xor(s, off, 32);
    if (t == 0) {
      atomicAdd(loss_acc, s);
      idxf_out[row] = (float)id;
    }
  }
}

// ----------------- finalize: loss scalar + perplexity -----------------
__global__ __launch_bounds__(1024)
void vq_finalize_kernel(const int* __restrict__ counts,
                        const float* __restrict__ loss_acc,
                        float* __restrict__ out_loss,
                        float* __restrict__ out_perp) {
  __shared__ float red[32];
  const int t = threadIdx.x;  // 0..1023, one per code
  const float p = (float)counts[t] * (1.0f / (float)VQ_BATCH);
  float s = -p * __logf(p + 1e-10f);
#pragma unroll
  for (int off = 16; off >= 1; off >>= 1) s += __shfl_xor(s, off, 32);
  if ((t & 31) == 0) red[t >> 5] = s;
  __syncthreads();
  if (t < 32) {
    float s2 = red[t];
#pragma unroll
    for (int off = 16; off >= 1; off >>= 1) s2 += __shfl_xor(s2, off, 32);
    if (t == 0) {
      out_perp[0] = __expf(s2);
      out_loss[0] = (1.0f + VQ_BETA) * loss_acc[0] *
                    (1.0f / ((float)VQ_BATCH * (float)VQ_DIM));
    }
  }
}

// ---------------------------------------------------------------------------
extern "C" void kernel_launch(void* const* d_in, const int* in_sizes, int n_in,
                              void* d_out, int out_size, void* d_ws, size_t ws_size,
                              hipStream_t stream) {
  (void)in_sizes; (void)n_in; (void)out_size; (void)ws_size;
  const float* z   = (const float*)d_in[0];           // [B, 256]
  const float* emb = (const float*)d_in[1];           // [1024, 256]

  // Workspace layout (all offsets 256B aligned; ~1.01 MB total)
  char* ws = (char*)d_ws;
  __bf16* embb     = (__bf16*)(ws + 0);              // 1024*256*2 = 524288 B
  float*  enorm    = (float*)(ws + 524288);          // 4096 B
  int*    counts   = (int*)(ws + 528384);            // 4096 B
  float*  loss_acc = (float*)(ws + 532480);          // 256 B
  int*    idx_ws   = (int*)(ws + 532736);            // 131072*4 = 524288 B

  // Output layout: z_q [B*256] | loss [1] | indices [B] | perplexity [1]
  float* out      = (float*)d_out;
  float* zq_out   = out;
  float* loss_out = out + (size_t)VQ_BATCH * VQ_DIM;
  float* idxf_out = loss_out + 1;
  float* perp_out = idxf_out + VQ_BATCH;

  vq_init_kernel<<<(VQ_NE + 255) / 256, 256, 0, stream>>>(counts, loss_acc);
  vq_prep_kernel<<<VQ_NE, 256, 0, stream>>>(emb, embb, enorm);
  // one wave per 16-row tile, 8 waves/block -> B/(16*8) = 1024 blocks
  vq_argmin_kernel<<<VQ_BATCH / 128, 256, VQ_LDS_BYTES, stream>>>(z, embb, enorm,
                                                                  idx_ws, counts);
  vq_gather_kernel<<<VQ_BATCH, 256, 0, stream>>>(z, emb, idx_ws, zq_out, idxf_out, loss_acc);
  vq_finalize_kernel<<<1, VQ_NE, 0, stream>>>(counts, loss_acc, loss_out, perp_out);
}
